// MultiheadCrossAttention_62491774157408
// MI455X (gfx1250) — matrix operand order repaired and checked by reading, not compiled, not run on hardware
//
#include <hip/hip_runtime.h>
#include <hip/hip_bf16.h>

typedef __attribute__((ext_vector_type(16))) _Float16 v16h;
typedef __attribute__((ext_vector_type(8)))  _Float16 v8h;
typedef __attribute__((ext_vector_type(8)))  float    v8f;

#define LOG2E 1.44269504088896340736f

// ---------------------------------------------------------------------------
// Fragment loaders (layouts per CDNA5 ISA 7.12.2, wave32)
// ---------------------------------------------------------------------------

// A fragment: 16x32 f16, row-major source with row stride lda (elements).
// lane: half = lane>>4, m = lane&15.
// regs 0-3 = K half*8..half*8+7 ; regs 4-7 = K 16+half*8..+7
__device__ __forceinline__ v16h load_a_frag(const _Float16* base, int lda, int lane) {
    int half = lane >> 4;
    int m    = lane & 15;
    const _Float16* p = base + (size_t)m * lda + half * 8;
    union { v16h v; v8h h[2]; } f;
    f.h[0] = *(const v8h*)(p);
    f.h[1] = *(const v8h*)(p + 16);
    return f.v;
}

// B fragment: 32x16 f16 from transposed (N-major) storage Bt[n][k], row
// stride ldb. lane column n = lane&15; regs 0-3 = K half*8..half*8+7,
// regs 4-7 = K 16+half*8..+7 (same order as A).
__device__ __forceinline__ v16h load_b_frag(const _Float16* baseT, int ldb, int lane) {
    int half = lane >> 4;
    int n    = lane & 15;
    const _Float16* p = baseT + (size_t)n * ldb + half * 8;
    union { v16h v; v8h h[2]; } f;
    f.h[0] = *(const v8h*)(p);
    f.h[1] = *(const v8h*)(p + 16);
    return f.v;
}

__device__ __forceinline__ v8f wmma_f16(v16h a, v16h b, v8f c) {
    return __builtin_amdgcn_wmma_f32_16x16x32_f16(false, a, false, b, (short)0, c, false, false);
}

__device__ __forceinline__ float rmax16(float v) {
    v = fmaxf(v, __shfl_xor(v, 1, 32));
    v = fmaxf(v, __shfl_xor(v, 2, 32));
    v = fmaxf(v, __shfl_xor(v, 4, 32));
    v = fmaxf(v, __shfl_xor(v, 8, 32));
    return v;
}
__device__ __forceinline__ float rsum16(float v) {
    v += __shfl_xor(v, 1, 32);
    v += __shfl_xor(v, 2, 32);
    v += __shfl_xor(v, 4, 32);
    v += __shfl_xor(v, 8, 32);
    return v;
}

// ---------------------------------------------------------------------------
// gfx1250 async copy: 8KB contiguous global -> LDS, 16 x b128 per lane.
// VDST VGPR carries the LDS byte offset (generic LDS addr low 32 bits).
// Tracked by ASYNCcnt.
// ---------------------------------------------------------------------------
__device__ __forceinline__ void async_copy_8k(unsigned lds_off, const _Float16* gsrc, int lane) {
    const char* g = (const char*)gsrc + (size_t)lane * 16;
    unsigned l = lds_off + (unsigned)lane * 16;
#pragma unroll
    for (int j = 0; j < 16; ++j) {
        asm volatile("global_load_async_to_lds_b128 %0, %1, off"
                     :: "v"(l + (unsigned)j * 512), "v"(g + (size_t)j * 512)
                     : "memory");
    }
}
__device__ __forceinline__ void wait_async_16() {
    asm volatile("s_wait_asynccnt 0x10" ::: "memory");
}
__device__ __forceinline__ void wait_async_0() {
    asm volatile("s_wait_asynccnt 0x0" ::: "memory");
}

// ---------------------------------------------------------------------------
// Conversion kernels
// ---------------------------------------------------------------------------
__global__ void cast_f32_f16(const float* __restrict__ in, _Float16* __restrict__ out, int n) {
    int i = blockIdx.x * blockDim.x + threadIdx.x;
    if (i < n) out[i] = (_Float16)in[i];
}

// in: K x N (row-major) fp32 -> out: N x K (row-major) f16  (out[n][k] = in[k][n])
__global__ void transpose_cast(const float* __restrict__ in, _Float16* __restrict__ out, int K, int N) {
    int i = blockIdx.x * blockDim.x + threadIdx.x;  // output linear index
    int n = i / K;
    int k = i - n * K;
    out[(size_t)n * K + k] = (_Float16)in[(size_t)k * N + n];
}

// ---------------------------------------------------------------------------
// GEMM mainloop: one wave computes a 32(M) x 64(N) tile.
// Alternating double-buffered fragments, unroll-by-2 over k (step 64),
// peeled epilogue -> no register copies, no branches in steady state.
// A row-major [.. x K], Bt N-major [N x K]. K must be a multiple of 64.
// ---------------------------------------------------------------------------
__device__ __forceinline__ void gemm_tile32(const _Float16* __restrict__ A, int lda,
                                            const _Float16* __restrict__ Bt, int ldb,
                                            int K, int lane, v8f c[2][4]) {
    v16h a0[2], a1[2], b0[4], b1[4];

    a0[0] = load_a_frag(A, lda, lane);
    a0[1] = load_a_frag(A + (size_t)16 * lda, lda, lane);
#pragma unroll
    for (int t = 0; t < 4; ++t) b0[t] = load_b_frag(Bt + (size_t)t * 16 * ldb, ldb, lane);
    a1[0] = load_a_frag(A + 32, lda, lane);
    a1[1] = load_a_frag(A + (size_t)16 * lda + 32, lda, lane);
#pragma unroll
    for (int t = 0; t < 4; ++t) b1[t] = load_b_frag(Bt + (size_t)t * 16 * ldb + 32, ldb, lane);

    for (int k = 64; k < K; k += 64) {
        // consume buffer 0 (data k-64), refill with data k
#pragma unroll
        for (int t = 0; t < 4; ++t) c[0][t] = wmma_f16(a0[0], b0[t], c[0][t]);
#pragma unroll
        for (int t = 0; t < 4; ++t) c[1][t] = wmma_f16(a0[1], b0[t], c[1][t]);
        __builtin_prefetch(A + k + 64 + (size_t)(lane & 15) * lda, 0, 1);  // global_prefetch_b8
        a0[0] = load_a_frag(A + k, lda, lane);
        a0[1] = load_a_frag(A + (size_t)16 * lda + k, lda, lane);
#pragma unroll
        for (int t = 0; t < 4; ++t) b0[t] = load_b_frag(Bt + (size_t)t * 16 * ldb + k, ldb, lane);

        // consume buffer 1 (data k-32), refill with data k+32
#pragma unroll
        for (int t = 0; t < 4; ++t) c[0][t] = wmma_f16(a1[0], b1[t], c[0][t]);
#pragma unroll
        for (int t = 0; t < 4; ++t) c[1][t] = wmma_f16(a1[1], b1[t], c[1][t]);
        a1[0] = load_a_frag(A + k + 32, lda, lane);
        a1[1] = load_a_frag(A + (size_t)16 * lda + k + 32, lda, lane);
#pragma unroll
        for (int t = 0; t < 4; ++t) b1[t] = load_b_frag(Bt + (size_t)t * 16 * ldb + k + 32, ldb, lane);
    }
    // epilogue: consume last two buffers (data K-64 and K-32)
#pragma unroll
    for (int t = 0; t < 4; ++t) c[0][t] = wmma_f16(a0[0], b0[t], c[0][t]);
#pragma unroll
    for (int t = 0; t < 4; ++t) c[1][t] = wmma_f16(a0[1], b0[t], c[1][t]);
#pragma unroll
    for (int t = 0; t < 4; ++t) c[0][t] = wmma_f16(a1[0], b1[t], c[0][t]);
#pragma unroll
    for (int t = 0; t < 4; ++t) c[1][t] = wmma_f16(a1[1], b1[t], c[1][t]);
}

// q = x @ Wq + bq, scaled by 1/sqrt(64); f16 output row-major [M x 1024]
__global__ __launch_bounds__(128) void gemm_q_kernel(const _Float16* __restrict__ A,
                                                     const _Float16* __restrict__ Bt,
                                                     const float* __restrict__ bias,
                                                     _Float16* __restrict__ Q, int K, int N) {
    int lane = threadIdx.x & 31;
    int wid  = threadIdx.x >> 5;
    int tileM = blockIdx.x;                    // 32 rows
    int tileN = (blockIdx.y * 4 + wid) * 64;
    const v8f vzero = {};
    v8f c[2][4] = {{vzero, vzero, vzero, vzero}, {vzero, vzero, vzero, vzero}};
    gemm_tile32(A + (size_t)tileM * 32 * K, K, Bt + (size_t)tileN * K, K, K, lane, c);
    int half = lane >> 4, n = lane & 15;
#pragma unroll
    for (int mh = 0; mh < 2; ++mh) {
#pragma unroll
        for (int t = 0; t < 4; ++t) {
            int col = tileN + t * 16 + n;
            float bv = bias[col];
#pragma unroll
            for (int r = 0; r < 8; ++r) {
                int row = tileM * 32 + mh * 16 + half * 8 + r;
                Q[(size_t)row * N + col] = (_Float16)((c[mh][t][r] + bv) * 0.125f);
            }
        }
    }
}

// kv = data @ Wkv + bkv; scatter K -> [b,h,key,d] and V -> [b,h,d,key]
__global__ __launch_bounds__(128) void gemm_kv_kernel(const _Float16* __restrict__ A,
                                                      const _Float16* __restrict__ Bt,
                                                      const float* __restrict__ bias,
                                                      _Float16* __restrict__ KB,
                                                      _Float16* __restrict__ VT, int K) {
    int lane = threadIdx.x & 31;
    int wid  = threadIdx.x >> 5;
    int tileM = blockIdx.x;                    // 32 rows
    int tileN = (blockIdx.y * 4 + wid) * 64;
    const v8f vzero = {};
    v8f c[2][4] = {{vzero, vzero, vzero, vzero}, {vzero, vzero, vzero, vzero}};
    gemm_tile32(A + (size_t)tileM * 32 * K, K, Bt + (size_t)tileN * K, K, K, lane, c);
    int half = lane >> 4, n = lane & 15;
#pragma unroll
    for (int mh = 0; mh < 2; ++mh) {
#pragma unroll
        for (int t = 0; t < 4; ++t) {
            int col = tileN + t * 16 + n;          // 0..2047
            int h = col >> 7;                      // head
            int inner = col & 127;                 // <64 -> K, >=64 -> V
            float bv = bias[col];
#pragma unroll
            for (int r = 0; r < 8; ++r) {
                int row = tileM * 32 + mh * 16 + half * 8 + r;   // 0..8191
                int b = row >> 12;
                int key = row & 4095;
                float val = c[mh][t][r] + bv;
                if (inner < 64) {
                    KB[(((size_t)(b * 16 + h) * 4096) + key) * 64 + inner] = (_Float16)val;
                } else {
                    VT[(((size_t)(b * 16 + h) * 64) + (inner - 64)) * 4096 + key] = (_Float16)val;
                }
            }
        }
    }
}

// final projection: out = O @ Wp + bp, fp32 output
__global__ __launch_bounds__(128) void gemm_out_kernel(const _Float16* __restrict__ A,
                                                       const _Float16* __restrict__ Bt,
                                                       const float* __restrict__ bias,
                                                       float* __restrict__ Out, int K, int N) {
    int lane = threadIdx.x & 31;
    int wid  = threadIdx.x >> 5;
    int tileM = blockIdx.x;                    // 32 rows
    int tileN = (blockIdx.y * 4 + wid) * 64;
    const v8f vzero = {};
    v8f c[2][4] = {{vzero, vzero, vzero, vzero}, {vzero, vzero, vzero, vzero}};
    gemm_tile32(A + (size_t)tileM * 32 * K, K, Bt + (size_t)tileN * K, K, K, lane, c);
    int half = lane >> 4, n = lane & 15;
#pragma unroll
    for (int mh = 0; mh < 2; ++mh) {
#pragma unroll
        for (int t = 0; t < 4; ++t) {
            int col = tileN + t * 16 + n;
            float bv = bias[col];
#pragma unroll
            for (int r = 0; r < 8; ++r) {
                int row = tileM * 32 + mh * 16 + half * 8 + r;
                Out[(size_t)row * N + col] = c[mh][t][r] + bv;
            }
        }
    }
}

// ---------------------------------------------------------------------------
// Flash attention: one wave per (b, h, 16-row q tile); online softmax over
// 4096 keys in 64-key blocks. K blocks (contiguous 8KB each) are double-
// buffered into LDS via gfx1250 async-to-LDS copies (ASYNCcnt).
// Per-wave LDS: 2 x 8KB K stage + 2KB P stage = 18KB; 4 waves -> 72KB.
// ---------------------------------------------------------------------------
__global__ __launch_bounds__(128) void attn_kernel(const _Float16* __restrict__ Q,
                                                   const _Float16* __restrict__ KB,
                                                   const _Float16* __restrict__ VT,
                                                   _Float16* __restrict__ OB) {
    __shared__ __align__(32) _Float16 smem[4][2 * 4096 + 1024];  // per wave: kst0,kst1,P
    int lane = threadIdx.x & 31;
    int wid  = threadIdx.x >> 5;
    int gw   = blockIdx.x * 4 + wid;       // 0..2047
    int b    = gw >> 10;                   // 1024 waves per batch (16 heads * 64 qtiles)
    int rem  = gw & 1023;
    int h    = rem >> 6;
    int qt   = rem & 63;

    const _Float16* Qb = Q + ((size_t)(b * 1024 + qt * 16)) * 1024 + h * 64;
    const _Float16* Kb = KB + (size_t)(b * 16 + h) * 4096 * 64;
    const _Float16* Vb = VT + (size_t)(b * 16 + h) * 64 * 4096;

    _Float16* kst0 = &smem[wid][0];
    _Float16* kst1 = &smem[wid][4096];
    _Float16* ldsP = &smem[wid][8192];
    unsigned kst_off[2] = { (unsigned)(uintptr_t)kst0, (unsigned)(uintptr_t)kst1 };
    const _Float16* kst[2] = { kst0, kst1 };

    v16h qa0 = load_a_frag(Qb, 1024, lane);
    v16h qa1 = load_a_frag(Qb + 32, 1024, lane);

    const v8f vzero = {};
    v8f o[4] = {vzero, vzero, vzero, vzero};
    float Mr[8], Lr[8];
#pragma unroll
    for (int r = 0; r < 8; ++r) { Mr[r] = -1e30f; Lr[r] = 0.0f; }

    int half = lane >> 4, nn = lane & 15;

    // prime the async pipeline with key block 0
    async_copy_8k(kst_off[0], Kb, lane);

    for (int kb = 0; kb < 64; ++kb) {
        // issue next block's copy, then wait for the current block's data
        if (kb + 1 < 64) {
            async_copy_8k(kst_off[(kb + 1) & 1], Kb + (size_t)(kb + 1) * 4096, lane);
            wait_async_16();
        } else {
            wait_async_0();
        }
        const _Float16* Kt = kst[kb & 1];

        // S = (q/sqrt(d)) . K^T  for 64 keys -> 4 tiles of 16x16 (reads LDS)
        v8f s[4] = {vzero, vzero, vzero, vzero};
#pragma unroll
        for (int t = 0; t < 4; ++t) {
            s[t] = wmma_f16(qa0, load_b_frag(Kt + (size_t)t * 16 * 64, 64, lane), s[t]);
            s[t] = wmma_f16(qa1, load_b_frag(Kt + (size_t)t * 16 * 64 + 32, 64, lane), s[t]);
        }
        // online softmax per row (rows m = half*8 + r)
#pragma unroll
        for (int r = 0; r < 8; ++r) {
            float m = fmaxf(fmaxf(s[0][r], s[1][r]), fmaxf(s[2][r], s[3][r]));
            m = rmax16(m);
            float mnew  = fmaxf(Mr[r], m);
            float alpha = exp2f((Mr[r] - mnew) * LOG2E);
            float psum  = 0.0f;
#pragma unroll
            for (int t = 0; t < 4; ++t) {
                float p = exp2f((s[t][r] - mnew) * LOG2E);
                psum += p;
                ldsP[(half * 8 + r) * 64 + t * 16 + nn] = (_Float16)p;
            }
            psum = rsum16(psum);
            Lr[r] = Lr[r] * alpha + psum;
            Mr[r] = mnew;
#pragma unroll
            for (int t = 0; t < 4; ++t) o[t][r] *= alpha;
        }
        // O += P @ V   (P 16x64 from LDS in A-layout, V^T[d][key] as B)
        v16h pa0 = load_a_frag(ldsP, 64, lane);
        v16h pa1 = load_a_frag(ldsP + 32, 64, lane);
        const _Float16* Vt0 = Vb + (size_t)kb * 64;
#pragma unroll
        for (int t = 0; t < 4; ++t) {
            o[t] = wmma_f16(pa0, load_b_frag(Vt0 + (size_t)t * 16 * 4096, 4096, lane), o[t]);
            o[t] = wmma_f16(pa1, load_b_frag(Vt0 + (size_t)t * 16 * 4096 + 32, 4096, lane), o[t]);
        }
    }

    // normalize and store O (f16) in [b, q, h*64+d] layout for the final GEMM
    _Float16* Ob = OB + ((size_t)(b * 1024 + qt * 16)) * 1024 + h * 64;
#pragma unroll
    for (int r = 0; r < 8; ++r) {
        float inv = 1.0f / Lr[r];
#pragma unroll
        for (int t = 0; t < 4; ++t) {
            Ob[(size_t)(half * 8 + r) * 1024 + t * 16 + nn] = (_Float16)(o[t][r] * inv);
        }
    }
}

// ---------------------------------------------------------------------------
// Launch
// ---------------------------------------------------------------------------
extern "C" void kernel_launch(void* const* d_in, const int* in_sizes, int n_in,
                              void* d_out, int out_size, void* d_ws, size_t ws_size,
                              hipStream_t stream) {
    const float* x    = (const float*)d_in[0];   // (2,1024,1024)
    const float* data = (const float*)d_in[1];   // (2,4096,1024)
    const float* Wq   = (const float*)d_in[2];   // (1024,1024)
    const float* bq   = (const float*)d_in[3];   // (1024)
    const float* Wkv  = (const float*)d_in[4];   // (1024,2048)
    const float* bkv  = (const float*)d_in[5];   // (2048)
    const float* Wp   = (const float*)d_in[6];   // (1024,1024)
    const float* bp   = (const float*)d_in[7];   // (1024)
    float* out = (float*)d_out;                  // (2,1024,1024)

    char* ws = (char*)d_ws;
    size_t off = 0;
    _Float16* XH   = (_Float16*)(ws + off); off += (size_t)2 * 1024 * 1024 * 2;      // x f16
    _Float16* DH   = (_Float16*)(ws + off); off += (size_t)2 * 4096 * 1024 * 2;      // data f16
    _Float16* WQT  = (_Float16*)(ws + off); off += (size_t)1024 * 1024 * 2;          // Wq^T f16
    _Float16* WKVT = (_Float16*)(ws + off); off += (size_t)2048 * 1024 * 2;          // Wkv^T f16
    _Float16* WPT  = (_Float16*)(ws + off); off += (size_t)1024 * 1024 * 2;          // Wp^T f16
    _Float16* Qb   = (_Float16*)(ws + off); off += (size_t)2 * 1024 * 1024 * 2;      // q (scaled)
    _Float16* KBf  = (_Float16*)(ws + off); off += (size_t)2 * 16 * 4096 * 64 * 2;   // K [b,h,n,d]
    _Float16* VTf  = (_Float16*)(ws + off); off += (size_t)2 * 16 * 64 * 4096 * 2;   // V^T [b,h,d,n]
    _Float16* OB   = (_Float16*)(ws + off); off += (size_t)2 * 1024 * 1024 * 2;      // attn out f16

    // 1) precision conversion + weight transposes
    cast_f32_f16<<<(2 * 1024 * 1024) / 256, 256, 0, stream>>>(x, XH, 2 * 1024 * 1024);
    cast_f32_f16<<<(2 * 4096 * 1024) / 256, 256, 0, stream>>>(data, DH, 2 * 4096 * 1024);
    transpose_cast<<<(1024 * 1024) / 256, 256, 0, stream>>>(Wq, WQT, 1024, 1024);
    transpose_cast<<<(1024 * 2048) / 256, 256, 0, stream>>>(Wkv, WKVT, 1024, 2048);
    transpose_cast<<<(1024 * 1024) / 256, 256, 0, stream>>>(Wp, WPT, 1024, 1024);

    // 2) q projection: M=2048, N=1024, K=1024 (scale 1/8 folded in)
    gemm_q_kernel<<<dim3(2048 / 32, 1024 / 256), 128, 0, stream>>>(XH, WQT, bq, Qb, 1024, 1024);

    // 3) kv projection: M=8192, N=2048, K=1024; scatter into K / V^T layouts
    gemm_kv_kernel<<<dim3(8192 / 32, 2048 / 256), 128, 0, stream>>>(DH, WKVT, bkv, KBf, VTf, 1024);

    // 4) flash attention: 2048 q-tiles, 4 waves per block
    attn_kernel<<<2048 / 4, 128, 0, stream>>>(Qb, KBf, VTf, OB);

    // 5) output projection: M=2048, N=1024, K=1024 -> fp32
    gemm_out_kernel<<<dim3(2048 / 32, 1024 / 256), 128, 0, stream>>>(OB, WPT, bp, out, 1024, 1024);
}
